// EntropyPatchEncoder_62612033241429
// MI455X (gfx1250) — matrix operand order repaired and checked by reading, not compiled, not run on hardware
//
#include <hip/hip_runtime.h>
#include <hip/hip_bf16.h>

#define B_ 4
#define T_ 8192
#define D_ 1024
#define H_ 512
#define K_PATCH 512
#define SEG 16
#define KTOT 3072   // 3 * D_

typedef __bf16 v16bf __attribute__((ext_vector_type(16)));
typedef __bf16 v2bf  __attribute__((ext_vector_type(2)));
typedef float  v8f   __attribute__((ext_vector_type(8)));

union AFrag { v16bf v; unsigned int u[8]; };

// Native hardware f32->bf16 conversion (RNE), packed pair -> one u32.
__device__ __forceinline__ unsigned int pack2_bf16(float a, float b) {
    v2bf v;
    v[0] = (__bf16)a;
    v[1] = (__bf16)b;
    return __builtin_bit_cast(unsigned int, v);
}
__device__ __forceinline__ unsigned short f32_to_bf16(float f) {
    __bf16 h = (__bf16)f;
    return __builtin_bit_cast(unsigned short, h);
}
__device__ __forceinline__ float bf16_to_f32(unsigned short h) {
    return __uint_as_float(((unsigned int)h) << 16);
}

// ---- Kernel 1: repack conv1_w (h, D, 3) f32 -> W2[o][k*1024+i] bf16 ----
__global__ void prep_w_kernel(const float* __restrict__ w1, unsigned short* __restrict__ W2) {
    int tid = blockIdx.x * blockDim.x + threadIdx.x;
    if (tid >= H_ * KTOT) return;
    int o   = tid / KTOT;
    int rem = tid - o * KTOT;
    int k   = rem >> 10;        // 0..2
    int i   = rem & 1023;
    W2[tid] = f32_to_bf16(w1[(size_t)o * KTOT + i * 3 + k]);
}

// ---- Kernel 2: conv1 as GEMM (bf16 WMMA, f32 accum) + exact GELU -> hcv bf16 ----
// Workgroup tile: M=128 (8 waves x 16), N=64 (4 accumulators/wave), K-chunk=64.
// Double-buffered LDS, software-pipelined: chunk k+1's weight tile streams in via
// global_load_async_to_lds_b128 (ASYNCcnt) and x tile via global loads while the
// 8 WMMAs of chunk k run; one barrier per chunk.
__global__ __launch_bounds__(256)
void conv1_gelu_kernel(const float* __restrict__ x, const unsigned short* __restrict__ W2,
                       const float* __restrict__ b1, unsigned short* __restrict__ hcv) {
    __shared__ unsigned short Alds[2][128 * 64];  // 2 x 16 KB weight tiles
    __shared__ unsigned short Blds[2][64 * 64];   // 2 x  8 KB activation tiles

    const int tbase  = blockIdx.x * 64;    // N tile (time)
    const int mblock = blockIdx.y * 128;   // M tile (out channels)
    const int b      = blockIdx.z;
    const int tid    = threadIdx.x;
    const int wv     = tid >> 5;           // wave id (wave32)
    const int lane   = tid & 31;
    const int n      = lane & 15;
    const int half   = lane >> 4;

    // Only the first and last time-blocks can touch the zero pad (uniform flag).
    const bool interior = (tbase != 0) && (tbase + 64 != T_);

    const unsigned ldsA0 = (unsigned)(size_t)(void*)&Alds[0][0];
    const unsigned ldsA1 = (unsigned)(size_t)(void*)&Alds[1][0];

    v8f acc[4] = {};

    // B-tile load mapping: thread -> (row t, 16-wide i segment)
    const int trow = tid >> 2;            // 0..63
    const int iseg = (tid & 3) << 4;      // 0,16,32,48

    auto issue_asyncA = [&](int kk, int buf) {
        const unsigned base = buf ? ldsA1 : ldsA0;
        #pragma unroll
        for (int r = 0; r < 4; ++r) {
            int ch  = tid + r * 256;
            int row = ch >> 3;
            int c8  = (ch & 7) << 3;
            unsigned long long ga = (unsigned long long)(size_t)
                (W2 + (size_t)(mblock + row) * KTOT + kk + c8);
            unsigned la = base + (unsigned)(ch << 4);
            asm volatile("global_load_async_to_lds_b128 %0, %1, off"
                         :: "v"(la), "v"(ga) : "memory");
        }
    };

    auto load_b_regs = [&](int kk, float4& v0, float4& v1, float4& v2, float4& v3) {
        const int kshift = kk >> 10;          // conv tap 0..2 (64 | 1024)
        const int ibase  = kk & 1023;
        const int tg     = tbase + trow + kshift - 1;
        const float4* src =
            (const float4*)(x + ((size_t)b * T_ + tg) * D_ + ibase + iseg);
        if (interior) {
            v0 = src[0]; v1 = src[1]; v2 = src[2]; v3 = src[3];
        } else {
            v0 = v1 = v2 = v3 = make_float4(0.f, 0.f, 0.f, 0.f);
            if (tg >= 0 && tg < T_) { v0 = src[0]; v1 = src[1]; v2 = src[2]; v3 = src[3]; }
        }
    };

    auto store_b = [&](int buf, float4 v0, float4 v1, float4 v2, float4 v3) {
        uint4* dst = (uint4*)&Blds[buf][trow * 64 + iseg];
        dst[0] = make_uint4(pack2_bf16(v0.x, v0.y), pack2_bf16(v0.z, v0.w),
                            pack2_bf16(v1.x, v1.y), pack2_bf16(v1.z, v1.w));
        dst[1] = make_uint4(pack2_bf16(v2.x, v2.y), pack2_bf16(v2.z, v2.w),
                            pack2_bf16(v3.x, v3.y), pack2_bf16(v3.z, v3.w));
    };

    // ---- Prologue: stage chunk 0 into buffer 0 ----
    float4 r0, r1, r2, r3;
    issue_asyncA(0, 0);
    load_b_regs(0, r0, r1, r2, r3);
    store_b(0, r0, r1, r2, r3);
    asm volatile("s_wait_asynccnt 0x0" ::: "memory");
    __syncthreads();

    // ---- Main pipelined loop: one barrier per 64-K chunk ----
    for (int kk = 0; kk < KTOT; kk += 64) {
        const int buf  = (kk >> 6) & 1;
        const int nbuf = buf ^ 1;
        const bool more = (kk + 64) < KTOT;

        if (more) {
            issue_asyncA(kk + 64, nbuf);               // weights -> other buffer
            load_b_regs(kk + 64, r0, r1, r2, r3);      // x tile  -> registers
        }

        // Compute: 2 K-steps x 4 N-subtiles, A fragment reused across N.
        const unsigned int* AU = (const unsigned int*)&Alds[buf][0];  // 32 u32/row
        const unsigned int* BU = (const unsigned int*)&Blds[buf][0];  // 32 u32/row
        const int arow = wv * 16 + n;
        #pragma unroll
        for (int s = 0; s < 2; ++s) {
            AFrag a;
            #pragma unroll
            for (int j = 0; j < 8; ++j) {
                int aidx = (j < 4 ? j : 8 + (j - 4)) + half * 4;  // ISA 16-bit A layout
                a.u[j] = AU[arow * 32 + s * 16 + aidx];
            }
            #pragma unroll
            for (int q = 0; q < 4; ++q) {
                AFrag bb;
                #pragma unroll
                for (int j = 0; j < 8; ++j)
                    bb.u[j] = BU[(q * 16 + n) * 32 + s * 16 + (j < 4 ? j : 8 + (j - 4)) + half * 4];
                acc[q] = __builtin_amdgcn_wmma_f32_16x16x32_bf16(
                    false, a.v, false, bb.v, (short)0, acc[q], false, false);
            }
        }

        if (more) {
            store_b(nbuf, r0, r1, r2, r3);             // converted x -> other buffer
            asm volatile("s_wait_asynccnt 0x0" ::: "memory");
        }
        __syncthreads();   // next buffer complete; current buffer free for overwrite
    }

    // ---- Epilogue: bias + exact GELU, pack bf16, 16B stores ----
    const int obase = mblock + wv * 16 + half * 8;
    float bias[8];
    #pragma unroll
    for (int j = 0; j < 8; ++j) bias[j] = b1[obase + j];
    #pragma unroll
    for (int q = 0; q < 4; ++q) {
        const int t = tbase + q * 16 + n;
        unsigned int packed[4];
        #pragma unroll
        for (int j = 0; j < 8; j += 2) {
            float y0 = acc[q][j]     + bias[j];
            float y1 = acc[q][j + 1] + bias[j + 1];
            y0 = 0.5f * y0 * (1.0f + erff(y0 * 0.70710678118654752f));
            y1 = 0.5f * y1 * (1.0f + erff(y1 * 0.70710678118654752f));
            packed[j >> 1] = pack2_bf16(y0, y1);
        }
        *(uint4*)(hcv + ((size_t)b * T_ + t) * H_ + obase) =
            make_uint4(packed[0], packed[1], packed[2], packed[3]);
    }
}

// ---- Kernel 3: conv2 (512 -> 1, k=3) -> logits (B, T). One wave per (b,t). ----
__global__ __launch_bounds__(256)
void conv2_logits_kernel(const unsigned short* __restrict__ hcv, const float* __restrict__ w2,
                         const float* __restrict__ b2, float* __restrict__ logits) {
    int gw   = blockIdx.x * 8 + (threadIdx.x >> 5);  // global wave = b*T + t
    int lane = threadIdx.x & 31;
    int b    = gw >> 13;
    int t    = gw & (T_ - 1);
    float acc = 0.f;
    #pragma unroll
    for (int k = 0; k < 3; ++k) {
        int tg = t + k - 1;
        if (tg < 0 || tg >= T_) continue;
        const unsigned short* row = hcv + ((size_t)b * T_ + tg) * H_;
        for (int o = lane; o < H_; o += 32)
            acc += bf16_to_f32(row[o]) * w2[o * 3 + k];
    }
    #pragma unroll
    for (int off = 16; off > 0; off >>= 1)
        acc += __shfl_xor(acc, off, 32);
    if (lane == 0) logits[(size_t)b * T_ + t] = acc + b2[0];
}

// ---- Kernel 4: per-(b,patch) softmax/centroid/entropy + gather + RMSNorm ----
__global__ __launch_bounds__(256)
void patch_kernel(const float* __restrict__ x, const float* __restrict__ logits,
                  const float* __restrict__ norm_w, float* __restrict__ out) {
    __shared__ float red[256];
    __shared__ int s_idx;
    const int bp = blockIdx.x;
    const int b  = bp >> 9;
    const int p  = bp & (K_PATCH - 1);

    if (threadIdx.x == 0) {
        const float* l = logits + (size_t)b * T_ + p * SEG;
        float ls[SEG];
        float mx = -1e30f;
        #pragma unroll
        for (int s = 0; s < SEG; ++s) { ls[s] = l[s]; mx = fmaxf(mx, ls[s]); }
        float z = 0.f;
        #pragma unroll
        for (int s = 0; s < SEG; ++s) { ls[s] = expf(ls[s] - mx); z += ls[s]; }
        float inv = 1.f / z;
        float bound = 0.f, ent = 0.f;
        #pragma unroll
        for (int s = 0; s < SEG; ++s) {
            float w = ls[s] * inv;
            bound += w * (float)(p * SEG + s);
            ent   -= w * logf(fmaxf(w, 1e-8f));
        }
        ent *= (1.0f / logf((float)SEG));
        int idx = (int)bound;                 // boundaries >= 0, trunc == floor
        idx = min(max(idx, 0), T_ - 1);
        s_idx = idx;
        out[(size_t)B_ * K_PATCH * D_ + (size_t)b * K_PATCH + p] = ent;
    }
    __syncthreads();

    const int idx = s_idx;
    const float4* r4 = (const float4*)(x + ((size_t)b * T_ + idx) * D_);
    float4 val = r4[threadIdx.x];             // 256 thr x 4 = 1024
    red[threadIdx.x] = val.x*val.x + val.y*val.y + val.z*val.z + val.w*val.w;
    __syncthreads();
    for (int s2 = 128; s2 > 0; s2 >>= 1) {
        if (threadIdx.x < s2) red[threadIdx.x] += red[threadIdx.x + s2];
        __syncthreads();
    }
    const float rms = rsqrtf(red[0] * (1.0f / D_) + 1e-6f);
    const float4 nw = ((const float4*)norm_w)[threadIdx.x];
    float4 ov = make_float4(val.x * rms * nw.x, val.y * rms * nw.y,
                            val.z * rms * nw.z, val.w * rms * nw.w);
    ((float4*)(out + ((size_t)b * K_PATCH + p) * D_))[threadIdx.x] = ov;
}

extern "C" void kernel_launch(void* const* d_in, const int* in_sizes, int n_in,
                              void* d_out, int out_size, void* d_ws, size_t ws_size,
                              hipStream_t stream) {
    (void)in_sizes; (void)n_in; (void)out_size; (void)ws_size;
    const float* x  = (const float*)d_in[0];
    const float* w1 = (const float*)d_in[1];
    const float* b1 = (const float*)d_in[2];
    const float* w2 = (const float*)d_in[3];
    const float* b2 = (const float*)d_in[4];
    const float* nw = (const float*)d_in[5];
    float* out = (float*)d_out;

    char* ws = (char*)d_ws;
    unsigned short* W2  = (unsigned short*)ws;                       // 3,145,728 B
    unsigned short* hcv = (unsigned short*)(ws + 3145728);           // 33,554,432 B
    float*          lg  = (float*)(ws + 3145728 + 33554432);         //   131,072 B

    prep_w_kernel<<<(H_ * KTOT) / 256, 256, 0, stream>>>(w1, W2);

    dim3 g2(T_ / 64, H_ / 128, B_);
    conv1_gelu_kernel<<<g2, 256, 0, stream>>>(x, W2, b1, hcv);

    conv2_logits_kernel<<<(B_ * T_) / 8, 256, 0, stream>>>(hcv, w2, b2, lg);

    patch_kernel<<<B_ * K_PATCH, 256, 0, stream>>>(x, lg, nw, out);
}
